// TreeEncoder_16458314678333
// MI455X (gfx1250) — matrix operand order repaired and checked by reading, not compiled, not run on hardware
//
#include <hip/hip_runtime.h>
#include <stdint.h>

typedef _Float16 h16;
typedef __attribute__((ext_vector_type(16))) _Float16 v16h;
typedef __attribute__((ext_vector_type(8)))  float    v8f;

typedef unsigned int u32x4 __attribute__((ext_vector_type(4)));
typedef int          i32x8 __attribute__((ext_vector_type(8)));
typedef int          i32x4 __attribute__((ext_vector_type(4)));

#define CONV_KS   18          // 576 / 32
#define SQ_KS     2           // 64  / 32
#define TOTAL_NODES 349525
#define CONV_PAN_ELEMS (4*CONV_KS*32*16)   // 36864 f16 per depth
#define SQ_PAN_ELEMS   (4*SQ_KS*32*16)     // 4096  f16

#if defined(__has_builtin)
# if __has_builtin(__builtin_amdgcn_tensor_load_to_lds) && __has_builtin(__builtin_amdgcn_s_wait_tensorcnt)
#  define USE_TDM 1
# endif
#endif
#ifndef USE_TDM
# define USE_TDM 0
#endif

union Frag {
  v16h     h;
  uint32_t u[8];
  h16      e[16];
};

// ---------- Morton helpers (match reference bit tricks) ----------
__device__ __forceinline__ uint32_t deint(uint32_t v) {
  v &= 0x55555555u;
  v = (v | (v >> 1)) & 0x33333333u;
  v = (v | (v >> 2)) & 0x0F0F0F0Fu;
  v = (v | (v >> 4)) & 0x00FF00FFu;
  v = (v | (v >> 8)) & 0x0000FFFFu;
  return v;
}
__device__ __forceinline__ uint32_t intl(uint32_t v) {
  v &= 0xFFFFu;
  v = (v | (v << 8)) & 0x00FF00FFu;
  v = (v | (v << 4)) & 0x0F0F0F0Fu;
  v = (v | (v << 2)) & 0x33333333u;
  v = (v | (v << 1)) & 0x55555555u;
  return v;
}

// B-fragment swizzle: element (k,n) -> flat f16 index inside panel.
// This code's contiguous k order (not AMD's table order): lane = n%16 + 16*((k%32)/16),
// within-lane slot holds 16 consecutive-k f16 (2 per VGPR).
__device__ __forceinline__ int frag_index(int k, int n, int ksteps) {
  int nt = n >> 4, nl = n & 15;
  int ks = k >> 5, kl = k & 31;
  int lane = nl + (kl & 16);
  return (((nt * ksteps + ks) * 32) + lane) * 16 + (kl & 15);
}

// A k_local, contiguous order matching B panels: vgpr v, lane-half hi -> k = 16*hi + 2*v
__device__ __forceinline__ int a_klocal(int v, int hi) {
  return (hi << 4) + (v << 1);
}

// Branchless positional-encoding component p of the 40-wide in_proj input.
// All paths are computed with selects (v_cndmask) — no EXEC-divergent branches.
// sin/cos unified via +pi/2 phase so each element costs one v_sin_f32.
__device__ __forceinline__ float posc(int p, float x, float y, float dn, float f) {
  int q = p - 4;                       // periodic part index (may be negative)
  int a = q / 12;                      // axis 0..2 for p in [4,40)
  int r = q % 12;                      // 0..5 sin freqs, 6..11 cos freqs
  int e = r - ((r >= 6) ? 6 : 0);      // octave 0..5
  float base = (a == 0) ? x : ((a == 1) ? y : dn);
  // exact 2^e without a transcendental: bit-assemble the fp32 exponent
  int ec = e < 0 ? 0 : (e > 5 ? 5 : e);
  float w = __int_as_float((ec + 127) << 23);
  float ang = base * 6.28318530717958647692f * w
            + ((r >= 6) ? 1.57079632679489661923f : 0.f);   // cos = sin(+pi/2)
  float per = __sinf(ang);
  float low = (p == 0) ? f : ((p == 1) ? x : ((p == 2) ? y : dn));
  float val = (p < 4) ? low : per;
  return (p >= 40) ? 0.f : val;
}

// ---------- weight prep: fp32 -> f16 panels in WMMA-B fragment order ----------
__global__ void __launch_bounds__(256)
prep_kernel(const float* __restrict__ ipW, const float* __restrict__ embW,
            const float* __restrict__ convW,
            h16* __restrict__ ipPan, h16* __restrict__ embPan, h16* __restrict__ convPan) {
  int idx = blockIdx.x * blockDim.x + threadIdx.x;
  if (idx < SQ_PAN_ELEMS) {                       // in_proj_W: 40x64 padded to 64x64
    int k = idx >> 6, n = idx & 63;
    float v = (k < 40) ? ipW[k * 64 + n] : 0.f;
    ipPan[frag_index(k, n, SQ_KS)] = (h16)v;
    return;
  }
  idx -= SQ_PAN_ELEMS;
  if (idx < 10 * SQ_PAN_ELEMS) {                  // emb_W: 10 x 64x64
    int d = idx >> 12, rem = idx & 4095;
    int k = rem >> 6, n = rem & 63;
    embPan[d * SQ_PAN_ELEMS + frag_index(k, n, SQ_KS)] = (h16)embW[d * 4096 + rem];
    return;
  }
  idx -= 10 * SQ_PAN_ELEMS;
  if (idx < 8 * CONV_PAN_ELEMS) {                 // conv_W: depths 1..8, 576x64 each
    int c = idx / CONV_PAN_ELEMS + 1, rem = idx % CONV_PAN_ELEMS;
    int k = rem >> 6, n = rem & 63;
    convPan[c * CONV_PAN_ELEMS + frag_index(k, n, CONV_KS)] = (h16)convW[c * CONV_PAN_ELEMS + rem];
  }
}

// ---------- in_proj: [f | pos_enc(39)] @ W(40x64) + b, per depth ----------
__global__ void __launch_bounds__(256)
inproj_kernel(const float* __restrict__ feat, const h16* __restrict__ pan,
              const float* __restrict__ bias, h16* __restrict__ hOut, int N, int depth) {
  int lane = threadIdx.x & 31, wv = threadIdx.x >> 5;
  int tile = blockIdx.x * 8 + wv;
  int tiles = (N + 15) >> 4;
  if (tile >= tiles) return;
  int hi = lane >> 4, ml = lane & 15;
  int node = tile * 16 + ml;
  int nodeC = node < N ? node : N - 1;
  float f = feat[nodeC];
  uint32_t key = (uint32_t)node;
  float res = (float)(1 << depth);
  float x  = ((float)deint(key) + 0.5f) / res;
  float y  = ((float)deint(key >> 1) + 0.5f) / res;
  float dn = (float)depth / 9.0f;

  v8f acc[4] = {};
#pragma unroll
  for (int ks = 0; ks < SQ_KS; ++ks) {
    Frag a;
#pragma unroll
    for (int v = 0; v < 8; ++v) {
      int k0 = ks * 32 + a_klocal(v, hi);
      a.e[2 * v]     = (h16)posc(k0, x, y, dn, f);
      a.e[2 * v + 1] = (h16)posc(k0 + 1, x, y, dn, f);
    }
#pragma unroll
    for (int t = 0; t < 4; ++t) {
      Frag b;
      b.h = *(const v16h*)(pan + ((t * SQ_KS + ks) * 32 + lane) * 16);
      acc[t] = __builtin_amdgcn_wmma_f32_16x16x32_f16(false, a.h, false, b.h,
                                                      (short)0, acc[t], false, false);
    }
  }
#pragma unroll
  for (int t = 0; t < 4; ++t) {
    int n = ml + t * 16;
    float bn = bias[n];
#pragma unroll
    for (int r = 0; r < 8; ++r) {
      int nd = tile * 16 + r + (hi << 3);
      if (nd < N) hOut[(size_t)nd * 64 + n] = (h16)(acc[t][r] + bn);
    }
  }
}

// ---------- pool children + add to parent's in_proj value ----------
__global__ void __launch_bounds__(256)
pool_kernel(const h16* __restrict__ hChild, const h16* __restrict__ hPar,
            h16* __restrict__ dst, int Nc, int zrow) {
  int idx = blockIdx.x * blockDim.x + threadIdx.x;
  int tot = (Nc + zrow) * 64;
  if (idx >= tot) return;
  int node = idx >> 6, ch = idx & 63;
  if (node >= Nc) { dst[idx] = (h16)0.f; return; }   // zero row for invalid neighbors
  const h16* c = hChild + ((size_t)node * 4) * 64 + ch;
  float p = 0.25f * ((float)c[0] + (float)c[64] + (float)c[128] + (float)c[192]);
  dst[idx] = (h16)((float)hPar[idx] + p);
}

// ---------- conv: relu(gather9(hs) @ W(576x64) + b), weights LDS-resident via TDM ----------
__global__ void __launch_bounds__(256)
conv_kernel(const h16* __restrict__ hs, const h16* __restrict__ pan,
            const float* __restrict__ bias, h16* __restrict__ hOut, int N, int depth) {
  extern __shared__ h16 Blds[];   // 36864 f16 = 73728 B swizzled weight panel

#if USE_TDM
  if (threadIdx.x < 32) {         // wave 0 issues one TDM DMA for the whole panel
    uint64_t ga = (uint64_t)(uintptr_t)pan;
    uint32_t lb = (uint32_t)(size_t)Blds;
    const uint32_t n64 = (CONV_PAN_ELEMS * 2u) / 8u;   // 9216 8-byte elements
    u32x4 g0; i32x8 g1; i32x4 g2; i32x4 g3; i32x8 g4;
    g0[0] = 1u;                                        // count=1, user mode
    g0[1] = lb;                                        // lds_addr
    g0[2] = (uint32_t)ga;                              // global_addr[31:0]
    g0[3] = (uint32_t)((ga >> 32) & 0x01FFFFFFu) | (2u << 30);  // addr[56:32] | type=2
    g1[0] = (int)(3u << 16);                           // data_size=8B, no multicast
    g1[1] = (int)(n64 << 16);                          // tensor_dim0[15:0]
    g1[2] = (int)(1u << 16);                           // tensor_dim1 = 1
    g1[3] = (int)(n64 << 16);                          // tile_dim0
    g1[4] = 1;                                         // tile_dim1 = 1
    g1[5] = (int)n64;                                  // tensor_dim0_stride
    g1[6] = 0; g1[7] = 0;
    g2[0] = g2[1] = g2[2] = g2[3] = 0;
    g3[0] = g3[1] = g3[2] = g3[3] = 0;
    g4[0] = g4[1] = g4[2] = g4[3] = g4[4] = g4[5] = g4[6] = g4[7] = 0;
    __builtin_amdgcn_tensor_load_to_lds(g0, g1, g2, g3, g4, 0);
    __builtin_amdgcn_s_wait_tensorcnt(0);
  }
#else
  {
    const uint4* s = (const uint4*)pan;
    uint4* dv = (uint4*)Blds;
    for (int i = threadIdx.x; i < CONV_PAN_ELEMS / 8; i += 256) dv[i] = s[i];
  }
#endif
  __syncthreads();

  int lane = threadIdx.x & 31, wv = threadIdx.x >> 5;
  int tile = blockIdx.x * 8 + wv;
  int tiles = (N + 15) >> 4;
  if (tile >= tiles) return;
  int hi = lane >> 4, ml = lane & 15;
  int node = tile * 16 + ml;

  // 3x3 Morton neighbor row byte-offsets (row N = zero row)
  uint32_t rowoff[9];
  {
    int res = 1 << depth;
    int ix = (int)deint((uint32_t)node), iy = (int)deint((uint32_t)node >> 1);
#pragma unroll
    for (int j = 0; j < 9; ++j) {
      int dx = j % 3 - 1, dy = j / 3 - 1;
      int nx = ix + dx, ny = iy + dy;
      bool ok = (node < N) && (nx >= 0) && (nx < res) && (ny >= 0) && (ny < res);
      uint32_t rk = intl((uint32_t)nx) | (intl((uint32_t)ny) << 1);
      rowoff[j] = (ok ? rk : (uint32_t)N) * 128u;   // 64 ch * 2B
    }
  }

  v8f acc[4] = {};
  const char* hsb = (const char*)hs;
#pragma unroll
  for (int ks = 0; ks < CONV_KS; ++ks) {
    const int j  = ks >> 1;            // neighbor index (K = j*64 + ch)
    const int cb = (ks & 1) << 5;      // channel base within neighbor
    const char* rp = hsb + rowoff[j];
    Frag a;
#pragma unroll
    for (int v = 0; v < 8; ++v) {
      int kl = a_klocal(v, hi);
      a.u[v] = *(const uint32_t*)(rp + ((cb + kl) << 1));
    }
#pragma unroll
    for (int t = 0; t < 4; ++t) {
      Frag b;
      b.h = *(const v16h*)(Blds + ((t * CONV_KS + ks) * 32 + lane) * 16);
      acc[t] = __builtin_amdgcn_wmma_f32_16x16x32_f16(false, a.h, false, b.h,
                                                      (short)0, acc[t], false, false);
    }
  }

#pragma unroll
  for (int t = 0; t < 4; ++t) {
    int n = ml + t * 16;
    float bn = bias[n];
#pragma unroll
    for (int r = 0; r < 8; ++r) {
      int nd = tile * 16 + r + (hi << 3);
      if (nd < N) {
        float v = acc[t][r] + bn;
        hOut[(size_t)nd * 64 + n] = (h16)(v > 0.f ? v : 0.f);
      }
    }
  }
}

// ---------- emb GEMM + LayerNorm + depth gain ----------
__global__ void __launch_bounds__(256)
emb_kernel(const h16* __restrict__ hIn, const h16* __restrict__ pan,
           const float* __restrict__ eb, const float* __restrict__ lg,
           const float* __restrict__ lb, const float* __restrict__ dgain,
           float* __restrict__ out, int N) {
  int lane = threadIdx.x & 31, wv = threadIdx.x >> 5;
  int tile = blockIdx.x * 8 + wv;
  int tiles = (N + 15) >> 4;
  if (tile >= tiles) return;
  int hi = lane >> 4, ml = lane & 15;
  const h16* rowp = hIn + ((size_t)tile * 16 + ml) * 64;

  v8f acc[4] = {};
#pragma unroll
  for (int ks = 0; ks < SQ_KS; ++ks) {
    Frag a;
#pragma unroll
    for (int v = 0; v < 8; ++v) {
      int k = ks * 32 + a_klocal(v, hi);
      a.u[v] = *(const uint32_t*)(rowp + k);
    }
#pragma unroll
    for (int t = 0; t < 4; ++t) {
      Frag b;
      b.h = *(const v16h*)(pan + ((t * SQ_KS + ks) * 32 + lane) * 16);
      acc[t] = __builtin_amdgcn_wmma_f32_16x16x32_f16(false, a.h, false, b.h,
                                                      (short)0, acc[t], false, false);
    }
  }

  float gain = dgain[0];
  float eb4[4], g4[4], b4[4];
#pragma unroll
  for (int t = 0; t < 4; ++t) {
    int n = ml + t * 16;
    eb4[t] = eb[n]; g4[t] = lg[n]; b4[t] = lb[n];
  }

#pragma unroll
  for (int r = 0; r < 8; ++r) {
    float z[4], s = 0.f, s2 = 0.f;
#pragma unroll
    for (int t = 0; t < 4; ++t) { z[t] = acc[t][r] + eb4[t]; s += z[t]; s2 += z[t] * z[t]; }
#pragma unroll
    for (int o = 1; o < 16; o <<= 1) {              // reduce over the 16-lane half
      s  += __shfl_xor(s,  o, 32);
      s2 += __shfl_xor(s2, o, 32);
    }
    float mu  = s  * 0.015625f;
    float var = s2 * 0.015625f - mu * mu;
    float inv = rsqrtf(var + 1e-5f);
    int nd = tile * 16 + r + (hi << 3);
    if (nd < N) {
#pragma unroll
      for (int t = 0; t < 4; ++t) {
        int n = ml + t * 16;
        out[(size_t)nd * 64 + n] = gain * ((z[t] - mu) * inv * g4[t] + b4[t]);
      }
    }
  }
}

extern "C" void kernel_launch(void* const* d_in, const int* in_sizes, int n_in,
                              void* d_out, int out_size, void* d_ws, size_t ws_size,
                              hipStream_t stream) {
  (void)in_sizes; (void)n_in; (void)out_size; (void)ws_size;
  const float* feat  = (const float*)d_in[0];
  const float* ipW   = (const float*)d_in[1];
  const float* ipB   = (const float*)d_in[2];
  const float* convW = (const float*)d_in[3];
  const float* convB = (const float*)d_in[4];
  const float* embW  = (const float*)d_in[5];
  const float* embB  = (const float*)d_in[6];
  const float* lnG   = (const float*)d_in[7];
  const float* lnB   = (const float*)d_in[8];
  const float* dg    = (const float*)d_in[9];
  float* out = (float*)d_out;

  static const long long OFF[11] = {0,1,5,21,85,341,1365,5461,21845,87381,349525};

  char* ws = (char*)d_ws;
  size_t off = 0;
  auto alloc = [&](size_t bytes) { char* p = ws + off; off = (off + bytes + 255) & ~(size_t)255; return p; };
  h16* hF      = (h16*)alloc((size_t)TOTAL_NODES * 64 * 2);   // f16 activations, all depths
  h16* hsBuf   = (h16*)alloc((size_t)(65536 + 1) * 64 * 2);   // pre-conv sum + zero row
  h16* ipPan   = (h16*)alloc((size_t)SQ_PAN_ELEMS * 2);
  h16* embPan  = (h16*)alloc((size_t)10 * SQ_PAN_ELEMS * 2);
  h16* convPan = (h16*)alloc((size_t)10 * CONV_PAN_ELEMS * 2);

  {
    int total = SQ_PAN_ELEMS + 10 * SQ_PAN_ELEMS + 8 * CONV_PAN_ELEMS;
    prep_kernel<<<(total + 255) / 256, 256, 0, stream>>>(ipW, embW, convW, ipPan, embPan, convPan);
  }

  for (int d = 0; d <= 9; ++d) {
    int N = 1 << (2 * d);
    int grid = (((N + 15) / 16) + 7) / 8;
    inproj_kernel<<<grid, 256, 0, stream>>>(feat + OFF[d], ipPan, ipB, hF + OFF[d] * 64, N, d);
  }

  for (int d = 9; d >= 1; --d) {
    int c = d - 1;
    int Nc = 1 << (2 * c);
    int zr = (c >= 1) ? 1 : 0;
    h16* dst = (c >= 1) ? hsBuf : hF;   // depth 0: no conv, write final h directly
    int tot = (Nc + zr) * 64;
    pool_kernel<<<(tot + 255) / 256, 256, 0, stream>>>(hF + OFF[d] * 64, hF + OFF[c] * 64, dst, Nc, zr);
    if (c >= 1) {
      int grid = (((Nc + 15) / 16) + 7) / 8;
      conv_kernel<<<grid, 256, CONV_PAN_ELEMS * 2, stream>>>(
          hsBuf, convPan + (size_t)c * CONV_PAN_ELEMS, convB + c * 64, hF + OFF[c] * 64, Nc, c);
    }
  }

  for (int d = 0; d <= 9; ++d) {
    int N = 1 << (2 * d);
    int grid = (((N + 15) / 16) + 7) / 8;
    emb_kernel<<<grid, 256, 0, stream>>>(hF + OFF[d] * 64, embPan + (size_t)d * SQ_PAN_ELEMS,
                                         embB + d * 64, lnG + d * 64, lnB + d * 64, dg + d,
                                         out + OFF[d] * 64, N);
  }
}
